// SelfAttention_23957327577489
// MI455X (gfx1250) — compile-verified, matrix operand order checked by reading, not run on hardware
//
#include <hip/hip_runtime.h>
#include <hip/hip_bf16.h>

#define BATCH 4
#define SEQ   2048
#define EMB   512
#define HEADS 8
#define HDIM  64

typedef __attribute__((ext_vector_type(16))) _Float16 v16h;
typedef __attribute__((ext_vector_type(8)))  _Float16 v8h;
typedef __attribute__((ext_vector_type(4)))  _Float16 v4h;
typedef __attribute__((ext_vector_type(8)))  float    v8f;
typedef __attribute__((ext_vector_type(4)))  float    v4f;

// ---------------------------------------------------------------------------
// Fragment loader for 16-bit WMMA A/B operands (CDNA5 16x32 / 32x16 layout).
// Per lane the fragment is two contiguous 8-half (16B) runs: k = kb + {0..7}
// and k = kb + {16..23}, where kb = (lane<16 ? 0 : 8).  Caller passes the
// pointer at element (row/col, kb).  Lowers to two global/ds _b128 loads.
// ---------------------------------------------------------------------------
static __device__ __forceinline__ v16h frag_load(const _Float16* p) {
  v8h lo = *(const v8h*)(p);
  v8h hi = *(const v8h*)(p + 16);
  v16h r;
#pragma unroll
  for (int i = 0; i < 8; ++i) { r[i] = lo[i]; r[i + 8] = hi[i]; }
  return r;
}

static __device__ __forceinline__ v8f wmma_f16(v16h a, v16h b, v8f c) {
  return __builtin_amdgcn_wmma_f32_16x16x32_f16(false, a, false, b,
                                                (short)0, c, false, false);
}

// ---------------------------------------------------------------------------
// fp32 -> f16 conversion (vectorized by 4)
// ---------------------------------------------------------------------------
__global__ void __launch_bounds__(256)
cvt_f32_f16(const float* __restrict__ in, _Float16* __restrict__ out, int n4) {
  int i = blockIdx.x * 256 + threadIdx.x;
  if (i < n4) {
    v4f v = ((const v4f*)in)[i];
    v4h o;
#pragma unroll
    for (int t = 0; t < 4; ++t) o[t] = (_Float16)v[t];
    ((v4h*)out)[i] = o;
  }
}

// ---------------------------------------------------------------------------
// Bit-pack the int32 mask: 32 int -> 1 u32.  67 MB -> 2 MB, read once.
// ---------------------------------------------------------------------------
__global__ void __launch_bounds__(256)
pack_mask(const int* __restrict__ mask, unsigned int* __restrict__ bits,
          int nwords) {
  int i = blockIdx.x * 256 + threadIdx.x;
  if (i < nwords) {
    const int4* p = (const int4*)(mask + (size_t)i * 32);
    unsigned int w = 0;
#pragma unroll
    for (int t = 0; t < 8; ++t) {
      int4 m = p[t];
      w |= (m.x != 0 ? 1u : 0u) << (4 * t + 0);
      w |= (m.y != 0 ? 1u : 0u) << (4 * t + 1);
      w |= (m.z != 0 ? 1u : 0u) << (4 * t + 2);
      w |= (m.w != 0 ? 1u : 0u) << (4 * t + 3);
    }
    bits[i] = w;
  }
}

// ---------------------------------------------------------------------------
// GEMM: Y[M,N] = A[M,K] @ W[N,K]^T + bias[N]   (torch Linear semantics)
// Block = 128 threads (4 waves); block tile 128(M) x 64(N); wave tile 32x64:
// two A fragments amortize every W fragment (8 WMMAs per 6 fragment loads).
// MODE 0: f16 row-major output (stride N)           -> q, k projections
// MODE 1: f16 head-transposed output [B,H,HDIM,SEQ] -> v projection
// MODE 2: f32 row-major output                      -> final out-projection
// ---------------------------------------------------------------------------
template <int MODE>
__global__ void __launch_bounds__(128)
gemm16(const _Float16* __restrict__ A, const _Float16* __restrict__ W,
       const float* __restrict__ bias, void* __restrict__ outp,
       int M, int N, int K) {
  const int lane = threadIdx.x & 31;
  const int wave = threadIdx.x >> 5;
  const int r    = lane & 15;          // row of A frag / column of B frag
  const int hb   = (lane >> 4) * 8;    // per-half k-run base; also C row base
  const int m0   = blockIdx.x * 128 + wave * 32;
  const int n0   = blockIdx.y * 64;

  const _Float16* arow0 = A + (size_t)(m0 + r) * K;
  const _Float16* arow1 = A + (size_t)(m0 + 16 + r) * K;
  const _Float16* wrow[4];
#pragma unroll
  for (int nt = 0; nt < 4; ++nt)
    wrow[nt] = W + (size_t)(n0 + nt * 16 + r) * K;

  v8f acc[2][4] = {};
  const int ksteps = K >> 5;
  for (int ks = 0; ks < ksteps; ++ks) {
    const int ko = ks * 32 + hb;
    v16h a0 = frag_load(arow0 + ko);
    v16h a1 = frag_load(arow1 + ko);
    __builtin_prefetch(arow0 + ko + 128, 0, 1);
    __builtin_prefetch(arow1 + ko + 128, 0, 1);
#pragma unroll
    for (int nt = 0; nt < 4; ++nt) {
      v16h b = frag_load(wrow[nt] + ko);
      acc[0][nt] = wmma_f16(a0, b, acc[0][nt]);
      acc[1][nt] = wmma_f16(a1, b, acc[1][nt]);
    }
  }

#pragma unroll
  for (int mt = 0; mt < 2; ++mt) {
    const int mrow = m0 + mt * 16 + hb;
#pragma unroll
    for (int nt = 0; nt < 4; ++nt) {
      const int col = n0 + nt * 16 + r;
      const float bv = bias[col];
      if (MODE == 0) {
        _Float16* o = (_Float16*)outp;
#pragma unroll
        for (int j = 0; j < 8; ++j)
          o[(size_t)(mrow + j) * N + col] = (_Float16)(acc[mt][nt][j] + bv);
      } else if (MODE == 1) {
        // [B,H,HDIM,SEQ]; lane holds 8 consecutive sequence positions of one
        // (h,d) row -> single 16B store.
        _Float16* o = (_Float16*)outp;
        const int bb = m0 / SEQ;
        const int s0 = mrow % SEQ;
        const int h = col >> 6, d = col & 63;
        v8h pk;
#pragma unroll
        for (int j = 0; j < 8; ++j) pk[j] = (_Float16)(acc[mt][nt][j] + bv);
        *(v8h*)(o + ((size_t)((bb * HEADS + h) * HDIM + d)) * SEQ + s0) = pk;
      } else {
        float* o = (float*)outp;
#pragma unroll
        for (int j = 0; j < 8; ++j)
          o[(size_t)(mrow + j) * N + col] = acc[mt][nt][j] + bv;
      }
    }
  }
}

// ---------------------------------------------------------------------------
// Flash attention: one block = 4 waves = 64 query rows of one (b,h).
// Each wave owns 16 q-rows: S-tile 16x64 -> bitmask/scale -> online softmax
// (shfl reductions within half-wave lane groups) -> P staged through wave-
// private LDS (C-layout -> A-layout) -> O += P @ V with V stored d-major.
// ---------------------------------------------------------------------------
__global__ void __launch_bounds__(128)
flash_attn(const _Float16* __restrict__ qh, const _Float16* __restrict__ kh,
           const _Float16* __restrict__ vt,
           const unsigned int* __restrict__ mbits,
           _Float16* __restrict__ oh) {
  __shared__ __align__(16) _Float16 Plds[4][16 * 64];

  const int lane = threadIdx.x & 31;
  const int wave = threadIdx.x >> 5;
  const int r    = lane & 15;
  const int hb   = (lane >> 4) * 8;
  const int b    = blockIdx.y / HEADS;
  const int h    = blockIdx.y % HEADS;
  const int q0   = blockIdx.x * 64 + wave * 16;

  const float scale = 0.04419417382415922f;   // 1/sqrt(512) (ref scales by E)
  const int mwpr = SEQ / 32;                  // mask words per row

  // Q fragments for this wave's 16 rows (d = 0..31 and 32..63), held in VGPRs
  const _Float16* qrow = qh + ((size_t)b * SEQ + q0 + r) * EMB + h * HDIM;
  v16h Qa0 = frag_load(qrow + hb);
  v16h Qa1 = frag_load(qrow + 32 + hb);

  v8f O[4] = {};
  float mrun[8], lrun[8];
#pragma unroll
  for (int j = 0; j < 8; ++j) { mrun[j] = -1e38f; lrun[j] = 0.f; }

  const _Float16* kb_ = kh + (size_t)b * SEQ * EMB + h * HDIM;
  const _Float16* vb_ = vt + ((size_t)(b * HEADS + h) * HDIM) * SEQ;
  const unsigned int* mb_ = mbits + ((size_t)b * SEQ + q0 + hb) * mwpr;

  for (int kt = 0; kt < SEQ / 64; ++kt) {
    const int k0 = kt * 64;

    // ---- S = Q @ K^T : 4 C-tiles of 16x16, reduction over d (2 wmma each)
    v8f Sc[4];
#pragma unroll
    for (int nt = 0; nt < 4; ++nt) {
      const _Float16* krow = kb_ + (size_t)(k0 + nt * 16 + r) * EMB;
      v16h b0 = frag_load(krow + hb);
      v16h b1 = frag_load(krow + 32 + hb);
      v8f c = {};
      c = wmma_f16(Qa0, b0, c);
      c = wmma_f16(Qa1, b1, c);
      Sc[nt] = c;
    }
    __builtin_prefetch(kb_ + (size_t)(k0 + 64 + r) * EMB, 0, 1);
    __builtin_prefetch(vb_ + (size_t)r * SEQ + k0 + 64, 0, 1);

    // ---- mask (pre-scale, as reference) + scale, via packed bits:
    // one aligned u64 per row covers this 64-key tile.
    unsigned long long mw[8];
#pragma unroll
    for (int j = 0; j < 8; ++j)
      mw[j] = *(const unsigned long long*)(mb_ + (size_t)j * mwpr + (k0 >> 5));
#pragma unroll
    for (int nt = 0; nt < 4; ++nt) {
      const int bitpos = nt * 16 + r;
#pragma unroll
      for (int j = 0; j < 8; ++j) {
        float s = Sc[nt][j];
        Sc[nt][j] = ((mw[j] >> bitpos) & 1ull) ? s * scale : -4.4e18f;
      }
    }

    // ---- online softmax (row stats replicated over the 16-lane half-group)
#pragma unroll
    for (int j = 0; j < 8; ++j) {
      float v = fmaxf(fmaxf(Sc[0][j], Sc[1][j]), fmaxf(Sc[2][j], Sc[3][j]));
#pragma unroll
      for (int off = 1; off < 16; off <<= 1)
        v = fmaxf(v, __shfl_xor(v, off, 32));
      const float mn = fmaxf(mrun[j], v);
      const float alpha = __expf(mrun[j] - mn);
      mrun[j] = mn;
      float sum = 0.f;
#pragma unroll
      for (int nt = 0; nt < 4; ++nt) {
        float p = __expf(Sc[nt][j] - mn);
        Sc[nt][j] = p;
        sum += p;
      }
#pragma unroll
      for (int off = 1; off < 16; off <<= 1)
        sum += __shfl_xor(sum, off, 32);
      lrun[j] = lrun[j] * alpha + sum;
#pragma unroll
      for (int dt = 0; dt < 4; ++dt) O[dt][j] *= alpha;
    }

    // ---- stage P (C layout -> row-major f16 in wave-private LDS)
    _Float16* pl = &Plds[wave][0];
#pragma unroll
    for (int nt = 0; nt < 4; ++nt)
#pragma unroll
      for (int j = 0; j < 8; ++j)
        pl[(hb + j) * 64 + nt * 16 + r] = (_Float16)Sc[nt][j];

    // ---- O += P @ V   (V is d-major: B fragments load contiguously)
#pragma unroll
    for (int step = 0; step < 2; ++step) {
      v16h a = frag_load(&Plds[wave][r * 64 + step * 32 + hb]);
#pragma unroll
      for (int dt = 0; dt < 4; ++dt) {
        const _Float16* vrow =
            vb_ + (size_t)(dt * 16 + r) * SEQ + k0 + step * 32 + hb;
        O[dt] = wmma_f16(a, frag_load(vrow), O[dt]);
      }
    }
  }

  // ---- normalize and emit attention output (f16, row-major [B*S, E])
  _Float16* orow = oh + ((size_t)b * SEQ + q0 + hb) * EMB + h * HDIM;
#pragma unroll
  for (int dt = 0; dt < 4; ++dt)
#pragma unroll
    for (int j = 0; j < 8; ++j)
      orow[(size_t)j * EMB + dt * 16 + r] = (_Float16)(O[dt][j] / lrun[j]);
}

// ---------------------------------------------------------------------------
extern "C" void kernel_launch(void* const* d_in, const int* in_sizes, int n_in,
                              void* d_out, int out_size, void* d_ws, size_t ws_size,
                              hipStream_t stream) {
  const float* key   = (const float*)d_in[0];
  const float* value = (const float*)d_in[1];
  const float* query = (const float*)d_in[2];
  const int*   mask  = (const int*)d_in[3];
  const float* Wq = (const float*)d_in[4];
  const float* bq = (const float*)d_in[5];
  const float* Wk = (const float*)d_in[6];
  const float* bk = (const float*)d_in[7];
  const float* Wv = (const float*)d_in[8];
  const float* bv = (const float*)d_in[9];
  const float* Wo = (const float*)d_in[10];
  const float* bo = (const float*)d_in[11];
  float* out = (float*)d_out;

  const size_t NX = (size_t)BATCH * SEQ * EMB;          // 4,194,304
  const size_t NW = (size_t)EMB * EMB;                  // 262,144
  const size_t NMW = (size_t)BATCH * SEQ * (SEQ / 32);  // 524,288 mask words

  char* w = (char*)d_ws;
  _Float16* xq  = (_Float16*)w; w += NX * 2;
  _Float16* xk  = (_Float16*)w; w += NX * 2;
  _Float16* xv  = (_Float16*)w; w += NX * 2;
  _Float16* wqh = (_Float16*)w; w += NW * 2;
  _Float16* wkh = (_Float16*)w; w += NW * 2;
  _Float16* wvh = (_Float16*)w; w += NW * 2;
  _Float16* woh = (_Float16*)w; w += NW * 2;
  _Float16* qh  = (_Float16*)w; w += NX * 2;
  _Float16* kh  = (_Float16*)w; w += NX * 2;
  _Float16* vt  = (_Float16*)w; w += NX * 2;   // [B,H,HDIM,SEQ]
  unsigned int* mbits = (unsigned int*)w; w += NMW * 4;
  _Float16* oh  = xq;  // xq is dead after the q projection; reuse for attn out

  const int M = BATCH * SEQ;  // 8192
  const int gx = (int)((NX / 4 + 255) / 256);
  const int gw = (int)((NW / 4 + 255) / 256);
  cvt_f32_f16<<<gx, 256, 0, stream>>>(query, xq, (int)(NX / 4));
  cvt_f32_f16<<<gx, 256, 0, stream>>>(key,   xk, (int)(NX / 4));
  cvt_f32_f16<<<gx, 256, 0, stream>>>(value, xv, (int)(NX / 4));
  cvt_f32_f16<<<gw, 256, 0, stream>>>(Wq, wqh, (int)(NW / 4));
  cvt_f32_f16<<<gw, 256, 0, stream>>>(Wk, wkh, (int)(NW / 4));
  cvt_f32_f16<<<gw, 256, 0, stream>>>(Wv, wvh, (int)(NW / 4));
  cvt_f32_f16<<<gw, 256, 0, stream>>>(Wo, woh, (int)(NW / 4));
  pack_mask<<<(int)((NMW + 255) / 256), 256, 0, stream>>>(mask, mbits,
                                                          (int)NMW);

  dim3 gp(M / 128, EMB / 64);  // (64, 8)
  gemm16<0><<<gp, 128, 0, stream>>>(xq, wqh, bq, qh, M, EMB, EMB);
  gemm16<0><<<gp, 128, 0, stream>>>(xk, wkh, bk, kh, M, EMB, EMB);
  gemm16<1><<<gp, 128, 0, stream>>>(xv, wvh, bv, vt, M, EMB, EMB);

  flash_attn<<<dim3(SEQ / 64, BATCH * HEADS), 128, 0, stream>>>(qh, kh, vt,
                                                                mbits, oh);

  gemm16<2><<<gp, 128, 0, stream>>>(oh, woh, bo, out, M, EMB, EMB);
}
